// Transformer_60567628808742
// MI455X (gfx1250) — matrix operand order repaired and checked by reading, not compiled, not run on hardware
//
#include <hip/hip_runtime.h>
#include <hip/hip_bf16.h>

// ---------------------------------------------------------------------------
// Transformer forward for gfx1250 (MI455X). All GEMMs via v_wmma_f32_16x16x32_f16
// (fp32 global data, f16 LDS tiles double-buffered, fp32 accumulate).
// ---------------------------------------------------------------------------

typedef __attribute__((ext_vector_type(16))) _Float16 v16h;
typedef __attribute__((ext_vector_type(8)))  float    v8f;

#define D_MODEL 512
#define NLAYERS 6
#define HEADS   8
#define DKH     64
#define D_FF    2048
#define VOCAB   32000
#define BATCH   4
#define SEQ     512
#define BSROWS  (BATCH*SEQ)        // 2048

#define BM 128
#define BN 64
#define BK 32
#define NT 256                      // 8 waves of 32

union FragH { v16h h; uint4 u[2]; };

// ---------------------------------------------------------------------------
// Generic WMMA GEMM:  C[z] = scale * A[z](MxK) * B[z]  (+bias) (+relu)
//   A: row-major [M,K], row stride lda, per-z offset bz*aZb + hz*aZh
//   B: if !bKmajor: [N,K] row-major (i.e. op is A*B^T) row stride ldb
//      if  bKmajor: [K,N] row-major, row stride ldb
//   C: row-major [M,N], row stride ldc
// grid = (N/BN, M/BM, Z); z -> (bz = z/H, hz = z%H)
// Double-buffered LDS: stage tile i+1 while computing tile i.
// ---------------------------------------------------------------------------
__global__ __launch_bounds__(NT)
void gemm_wmma_kernel(const float* __restrict__ A, int lda, long long aZb, long long aZh,
                      const float* __restrict__ B, int ldb, long long bZb, long long bZh,
                      int bKmajor,
                      float* __restrict__ C, int ldc, long long cZb, long long cZh,
                      int K, const float* __restrict__ bias,
                      float scale, int relu, int H)
{
    __shared__ __align__(16) _Float16 As[2][BM][BK];   // 16 KB
    __shared__ __align__(16) _Float16 Bs[2][BN][BK];   //  8 KB

    const int tid  = threadIdx.x;
    const int z    = blockIdx.z;
    const int bz   = z / H;
    const int hz   = z % H;
    const int colBase = blockIdx.x * BN;

    const float* Ab = A + (size_t)bz * aZb + (size_t)hz * aZh
                        + (size_t)(blockIdx.y * BM) * lda;
    const float* Bb = B + (size_t)bz * bZb + (size_t)hz * bZh;

    const int lane = tid & 31;
    const int wv   = tid >> 5;       // wave id 0..7 -> M sub-slice
    const int half = lane >> 4;      // 0: lanes 0-15, 1: lanes 16-31
    const int m16  = lane & 15;

    // stage one 128x32 A tile + 64x32 B tile (fp32 -> f16) into LDS buffer `sb`
    auto stage = [&](int sb, int k0) {
#pragma unroll
        for (int i = 0; i < 4; ++i) {
            int f  = i * NT + tid;
            int r  = f >> 3;
            int c4 = (f & 7) << 2;
            float4 v = *(const float4*)(Ab + (size_t)r * lda + (k0 + c4));
            As[sb][r][c4 + 0] = (_Float16)v.x;
            As[sb][r][c4 + 1] = (_Float16)v.y;
            As[sb][r][c4 + 2] = (_Float16)v.z;
            As[sb][r][c4 + 3] = (_Float16)v.w;
        }
        if (!bKmajor) {
#pragma unroll
            for (int i = 0; i < 2; ++i) {
                int f  = i * NT + tid;
                int n  = f >> 3;
                int c4 = (f & 7) << 2;
                float4 v = *(const float4*)(Bb + (size_t)(colBase + n) * ldb + (k0 + c4));
                Bs[sb][n][c4 + 0] = (_Float16)v.x;
                Bs[sb][n][c4 + 1] = (_Float16)v.y;
                Bs[sb][n][c4 + 2] = (_Float16)v.z;
                Bs[sb][n][c4 + 3] = (_Float16)v.w;
            }
        } else {
#pragma unroll
            for (int i = 0; i < 2; ++i) {
                int f  = i * NT + tid;
                int kk = f >> 4;
                int n4 = (f & 15) << 2;
                float4 v = *(const float4*)(Bb + (size_t)(k0 + kk) * ldb + (colBase + n4));
                Bs[sb][n4 + 0][kk] = (_Float16)v.x;
                Bs[sb][n4 + 1][kk] = (_Float16)v.y;
                Bs[sb][n4 + 2][kk] = (_Float16)v.z;
                Bs[sb][n4 + 3][kk] = (_Float16)v.w;
            }
        }
    };

    v8f zero = {};
    v8f acc[4] = { zero, zero, zero, zero };

    const int nsteps = K / BK;
    stage(0, 0);
    __syncthreads();

    for (int it = 0; it < nsteps; ++it) {
        const int sb = it & 1;

        // ---- issue fragment loads from current buffer FIRST (in-order DS ->
        //      the pre-WMMA wait only needs to cover these, not the stores below)
        FragH af, bf[4];
        af.u[0] = *(const uint4*)&As[sb][wv * 16 + m16][half * 8];       // K = h*8..h*8+7
        af.u[1] = *(const uint4*)&As[sb][wv * 16 + m16][half * 8 + 16];  // K = h*8+16..+23
#pragma unroll
        for (int t = 0; t < 4; ++t) {
            bf[t].u[0] = *(const uint4*)&Bs[sb][t * 16 + m16][half * 8];
            bf[t].u[1] = *(const uint4*)&Bs[sb][t * 16 + m16][half * 8 + 16];
        }

        // ---- overlap: stage next K-tile into the other buffer while computing
        if (it + 1 < nsteps) {
            stage(sb ^ 1, (it + 1) * BK);
            if (it + 2 < nsteps)   // prefetch tile after next into cache
                __builtin_prefetch(Ab + (size_t)(tid >> 1) * lda + (it + 2) * BK, 0, 1);
        }

        // ---- 4 back-to-back WMMAs on the matrix pipe ----
#pragma unroll
        for (int t = 0; t < 4; ++t) {
            acc[t] = __builtin_amdgcn_wmma_f32_16x16x32_f16(
                         false, af.h, false, bf[t].h, (short)0, acc[t], false, false);
        }
        __syncthreads();
    }

    // ---- epilogue: C layout lane m16 = N col, VGPR v -> M = v + 8*half ----
    float* Cb = C + (size_t)bz * cZb + (size_t)hz * cZh
                  + (size_t)(blockIdx.y * BM + wv * 16) * ldc + colBase;
#pragma unroll
    for (int t = 0; t < 4; ++t) {
#pragma unroll
        for (int vi = 0; vi < 8; ++vi) {
            int mrow = vi + half * 8;
            int col  = t * 16 + m16;
            float v = acc[t][vi] * scale;
            if (bias) v += bias[colBase + col];
            if (relu) v = fmaxf(v, 0.0f);
            Cb[(size_t)mrow * ldc + col] = v;
        }
    }
}

// ---------------------------------------------------------------------------
// Embedding + sinusoidal positional encoding. grid = BSROWS, 256 threads.
// ---------------------------------------------------------------------------
__global__ __launch_bounds__(256)
void embed_kernel(const int* __restrict__ tokens, const float* __restrict__ table,
                  float* __restrict__ out)
{
    int r = blockIdx.x;                 // b*SEQ + s
    int s = r % SEQ;
    int tok = tokens[r];
    const float neg_ln = -9.2103403719761836f / (float)D_MODEL;  // -ln(10000)/D
#pragma unroll
    for (int i = 0; i < 2; ++i) {
        int d = threadIdx.x + i * 256;
        float div = __expf(neg_ln * (float)(d & ~1));
        float ang = (float)s * div;
        float pe  = (d & 1) ? __cosf(ang) : __sinf(ang);
        out[(size_t)r * D_MODEL + d] = table[(size_t)tok * D_MODEL + d] + pe;
    }
}

// ---------------------------------------------------------------------------
// Masked softmax over rows of scores [B*H*S, S]. mode 0:none 1:src(b,j) 2:tgt(b,i,j)
// ---------------------------------------------------------------------------
__global__ __launch_bounds__(256)
void softmax_mask_kernel(float* __restrict__ scores, const int* __restrict__ mask, int mode)
{
    __shared__ float red[256];
    int r = blockIdx.x;                 // (b*H + h)*S + i
    int b = r / (HEADS * SEQ);
    int i = r % SEQ;
    int tid = threadIdx.x;
    float* row = scores + (size_t)r * SEQ;

    float v0 = row[tid];
    float v1 = row[tid + 256];
    if (mode == 1) {
        const int* m = mask + (size_t)b * SEQ;          // [B,1,1,S]
        if (m[tid]       == 0) v0 = -1e9f;
        if (m[tid + 256] == 0) v1 = -1e9f;
    } else if (mode == 2) {
        const int* m = mask + (size_t)b * SEQ * SEQ + (size_t)i * SEQ;  // [B,1,S,S]
        if (m[tid]       == 0) v0 = -1e9f;
        if (m[tid + 256] == 0) v1 = -1e9f;
    }
    float mx = fmaxf(v0, v1);
    red[tid] = mx; __syncthreads();
    for (int s = 128; s > 0; s >>= 1) {
        if (tid < s) red[tid] = fmaxf(red[tid], red[tid + s]);
        __syncthreads();
    }
    mx = red[0]; __syncthreads();
    float e0 = __expf(v0 - mx), e1 = __expf(v1 - mx);
    red[tid] = e0 + e1; __syncthreads();
    for (int s = 128; s > 0; s >>= 1) {
        if (tid < s) red[tid] += red[tid + s];
        __syncthreads();
    }
    float inv = 1.0f / red[0];
    row[tid]       = e0 * inv;
    row[tid + 256] = e1 * inv;
}

// ---------------------------------------------------------------------------
// x = alpha * (t - mean) / sqrt(std_unbiased + eps) + beta, t = x + resid.
// Faithful to the reference quirk: divide by sqrt(std + eps), std ddof=1.
// ---------------------------------------------------------------------------
__global__ __launch_bounds__(256)
void add_layernorm_kernel(float* __restrict__ x, const float* __restrict__ resid,
                          const float* __restrict__ alpha, const float* __restrict__ beta)
{
    __shared__ float red[256];
    int row = blockIdx.x;
    int tid = threadIdx.x;
    float* xr = x + (size_t)row * D_MODEL;
    const float* rr = resid + (size_t)row * D_MODEL;

    float t0 = xr[tid]       + rr[tid];
    float t1 = xr[tid + 256] + rr[tid + 256];

    red[tid] = t0 + t1; __syncthreads();
    for (int s = 128; s > 0; s >>= 1) {
        if (tid < s) red[tid] += red[tid + s];
        __syncthreads();
    }
    float mean = red[0] * (1.0f / (float)D_MODEL);
    __syncthreads();

    float d0 = t0 - mean, d1 = t1 - mean;
    red[tid] = d0 * d0 + d1 * d1; __syncthreads();
    for (int s = 128; s > 0; s >>= 1) {
        if (tid < s) red[tid] += red[tid + s];
        __syncthreads();
    }
    float var = red[0] * (1.0f / (float)(D_MODEL - 1));   // unbiased
    float stdv = sqrtf(fmaxf(var, 0.0f));
    float inv = 1.0f / sqrtf(stdv + 1e-6f);
    xr[tid]       = alpha[tid]       * d0 * inv + beta[tid];
    xr[tid + 256] = alpha[tid + 256] * d1 * inv + beta[tid + 256];
}

// ---------------------------------------------------------------------------
// Host-side orchestration
// ---------------------------------------------------------------------------
struct MHAW { const float *Wq,*bq,*Wk,*bk,*Wv,*bv,*Wo,*bo; };
struct FFW  { const float *W1,*b1,*W2,*b2; };

static void launch_gemm(hipStream_t st,
                        const float* A, int lda, long long aZb, long long aZh,
                        const float* B, int ldb, long long bZb, long long bZh, int bKmaj,
                        float* C, int ldc, long long cZb, long long cZh,
                        int M, int N, int K, const float* bias,
                        float scale, int relu, int Z, int H)
{
    dim3 grid(N / BN, M / BM, Z);
    gemm_wmma_kernel<<<grid, dim3(NT), 0, st>>>(A, lda, aZb, aZh,
                                                B, ldb, bZb, bZh, bKmaj,
                                                C, ldc, cZb, cZh,
                                                K, bias, scale, relu, H);
}

static void run_mha(hipStream_t st, const float* qin, const float* kvin, const MHAW& p,
                    const int* mask, int mode,
                    float* q, float* k, float* v, float* scores, float* t1, float* t2)
{
    const long long SD  = (long long)SEQ * D_MODEL;
    const long long SS  = (long long)SEQ * SEQ;
    const long long HSS = (long long)HEADS * SS;

    // projections
    launch_gemm(st, qin, D_MODEL,0,0,  p.Wq, D_MODEL,0,0,0,  q, D_MODEL,0,0,
                BSROWS, D_MODEL, D_MODEL, p.bq, 1.0f, 0, 1, 1);
    launch_gemm(st, kvin,D_MODEL,0,0,  p.Wk, D_MODEL,0,0,0,  k, D_MODEL,0,0,
                BSROWS, D_MODEL, D_MODEL, p.bk, 1.0f, 0, 1, 1);
    launch_gemm(st, kvin,D_MODEL,0,0,  p.Wv, D_MODEL,0,0,0,  v, D_MODEL,0,0,
                BSROWS, D_MODEL, D_MODEL, p.bv, 1.0f, 0, 1, 1);

    // scores[b,h] = (1/8) * Q_bh (SxDk) * K_bh^T  -> [B*H, S, S]
    launch_gemm(st, q, D_MODEL, SD, DKH,
                    k, D_MODEL, SD, DKH, 0,
                    scores, SEQ, HSS, SS,
                    SEQ, SEQ, DKH, nullptr, 0.125f, 0, BATCH * HEADS, HEADS);

    softmax_mask_kernel<<<dim3(BATCH * HEADS * SEQ), dim3(256), 0, st>>>(scores, mask, mode);

    // t1[b,:,h*64+d] = attn_bh (SxS) * V_bh (SxDk)   (B is K-major here)
    launch_gemm(st, scores, SEQ, HSS, SS,
                    v, D_MODEL, SD, DKH, 1,
                    t1, D_MODEL, SD, DKH,
                    SEQ, DKH, SEQ, nullptr, 1.0f, 0, BATCH * HEADS, HEADS);

    // output projection
    launch_gemm(st, t1, D_MODEL,0,0,  p.Wo, D_MODEL,0,0,0,  t2, D_MODEL,0,0,
                BSROWS, D_MODEL, D_MODEL, p.bo, 1.0f, 0, 1, 1);
}

extern "C" void kernel_launch(void* const* d_in, const int* in_sizes, int n_in,
                              void* d_out, int out_size, void* d_ws, size_t ws_size,
                              hipStream_t stream)
{
    (void)in_sizes; (void)n_in; (void)out_size; (void)ws_size;

    // ---- inputs (setup_inputs dict order, params flattened in insertion order) ----
    const int* src      = (const int*)d_in[0];
    const int* trg      = (const int*)d_in[1];
    const int* src_mask = (const int*)d_in[2];
    const int* tgt_mask = (const int*)d_in[3];

    int idx = 4;
    auto nf = [&]() { return (const float*)d_in[idx++]; };

    const float* enc_embed = nf();
    const float* dec_embed = nf();

    struct EncL { MHAW attn; FFW ff; const float *l1a,*l1b,*l2a,*l2b; };
    struct DecL { MHAW a1, a2; FFW ff; const float *l1a,*l1b,*l2a,*l2b,*l3a,*l3b; };
    EncL enc[NLAYERS];
    DecL dec[NLAYERS];

    auto read_mha = [&](MHAW& m) {
        m.Wq = nf(); m.bq = nf(); m.Wk = nf(); m.bk = nf();
        m.Wv = nf(); m.bv = nf(); m.Wo = nf(); m.bo = nf();
    };
    auto read_ff = [&](FFW& f) { f.W1 = nf(); f.b1 = nf(); f.W2 = nf(); f.b2 = nf(); };

    for (int l = 0; l < NLAYERS; ++l) {
        read_mha(enc[l].attn); read_ff(enc[l].ff);
        enc[l].l1a = nf(); enc[l].l1b = nf(); enc[l].l2a = nf(); enc[l].l2b = nf();
    }
    for (int l = 0; l < NLAYERS; ++l) {
        read_mha(dec[l].a1); read_mha(dec[l].a2); read_ff(dec[l].ff);
        dec[l].l1a = nf(); dec[l].l1b = nf(); dec[l].l2a = nf(); dec[l].l2b = nf();
        dec[l].l3a = nf(); dec[l].l3b = nf();
    }
    const float* outW = nf();
    const float* outb = nf();

    // ---- workspace carve (floats) ----
    float* w = (float*)d_ws;
    const size_t XD = (size_t)BSROWS * D_MODEL;
    float* x      = w; w += XD;                              // encoder stream / enc_out
    float* y      = w; w += XD;                              // decoder stream
    float* qb     = w; w += XD;
    float* kb     = w; w += XD;
    float* vb     = w; w += XD;
    float* t1     = w; w += XD;
    float* t2     = w; w += XD;
    float* scores = w; w += (size_t)BATCH * HEADS * SEQ * SEQ;
    float* hid    = w; w += (size_t)BSROWS * D_FF;

    // ---- encoder ----
    embed_kernel<<<dim3(BSROWS), dim3(256), 0, stream>>>(src, enc_embed, x);
    for (int l = 0; l < NLAYERS; ++l) {
        run_mha(stream, x, x, enc[l].attn, src_mask, 1, qb, kb, vb, scores, t1, t2);
        add_layernorm_kernel<<<dim3(BSROWS), dim3(256), 0, stream>>>(x, t2, enc[l].l1a, enc[l].l1b);
        launch_gemm(stream, x, D_MODEL,0,0, enc[l].ff.W1, D_MODEL,0,0,0, hid, D_FF,0,0,
                    BSROWS, D_FF, D_MODEL, enc[l].ff.b1, 1.0f, 1, 1, 1);
        launch_gemm(stream, hid, D_FF,0,0, enc[l].ff.W2, D_FF,0,0,0, t2, D_MODEL,0,0,
                    BSROWS, D_MODEL, D_FF, enc[l].ff.b2, 1.0f, 0, 1, 1);
        add_layernorm_kernel<<<dim3(BSROWS), dim3(256), 0, stream>>>(x, t2, enc[l].l2a, enc[l].l2b);
    }

    // ---- decoder ----
    embed_kernel<<<dim3(BSROWS), dim3(256), 0, stream>>>(trg, dec_embed, y);
    for (int l = 0; l < NLAYERS; ++l) {
        run_mha(stream, y, y, dec[l].a1, tgt_mask, 2, qb, kb, vb, scores, t1, t2);
        add_layernorm_kernel<<<dim3(BSROWS), dim3(256), 0, stream>>>(y, t2, dec[l].l1a, dec[l].l1b);
        run_mha(stream, y, x, dec[l].a2, src_mask, 1, qb, kb, vb, scores, t1, t2);
        add_layernorm_kernel<<<dim3(BSROWS), dim3(256), 0, stream>>>(y, t2, dec[l].l2a, dec[l].l2b);
        launch_gemm(stream, y, D_MODEL,0,0, dec[l].ff.W1, D_MODEL,0,0,0, hid, D_FF,0,0,
                    BSROWS, D_FF, D_MODEL, dec[l].ff.b1, 1.0f, 1, 1, 1);
        launch_gemm(stream, hid, D_FF,0,0, dec[l].ff.W2, D_FF,0,0,0, t2, D_MODEL,0,0,
                    BSROWS, D_MODEL, D_FF, dec[l].ff.b2, 1.0f, 0, 1, 1);
        add_layernorm_kernel<<<dim3(BSROWS), dim3(256), 0, stream>>>(y, t2, dec[l].l3a, dec[l].l3b);
    }

    // ---- output projection -> logits [B*S, VOCAB] ----
    launch_gemm(stream, y, D_MODEL,0,0, outW, D_MODEL,0,0,0, (float*)d_out, VOCAB,0,0,
                BSROWS, VOCAB, D_MODEL, outb, 1.0f, 0, 1, 1);
}
